// FiLMRelationalMultiHeadAttentionMP_12403865551634
// MI455X (gfx1250) — matrix operand order repaired and checked by reading, not compiled, not run on hardware
//
#include <hip/hip_runtime.h>
#include <hip/hip_bf16.h>

typedef __attribute__((ext_vector_type(16))) _Float16 v16h;
typedef __attribute__((ext_vector_type(8)))  _Float16 v8h;
typedef __attribute__((ext_vector_type(8)))  float    v8f;
typedef __attribute__((ext_vector_type(4)))  int      v4i;

#define HID   128
#define HEADS 8
#define PHD   16

#if __has_builtin(__builtin_amdgcn_global_load_async_to_lds_b128) && \
    __has_builtin(__builtin_amdgcn_s_wait_asynccnt)
#define HAVE_ASYNC_LDS 1
typedef __attribute__((address_space(1))) v4i* gptr_v4i;
typedef __attribute__((address_space(3))) v4i* lptr_v4i;
#endif

__device__ __forceinline__ float atomAddF(float* p, float v) {
  return __hip_atomic_fetch_add(p, v, __ATOMIC_RELAXED, __HIP_MEMORY_SCOPE_AGENT);
}
// order-preserving float <-> uint encoding for atomicMax-based segment max
__device__ __forceinline__ unsigned ordEnc(float f) {
  unsigned u = __float_as_uint(f);
  return (u & 0x80000000u) ? ~u : (u | 0x80000000u);
}
__device__ __forceinline__ float ordDec(unsigned u) {
  unsigned v = (u & 0x80000000u) ? (u & 0x7FFFFFFFu) : ~u;
  return __uint_as_float(v);
}

// Butterfly sum across each 16-lane half using ds_swizzle_b32 XOR mode
// (offset = xor_mask<<10 | and_mask 0x1f ; CDNA5 ISA SWAPX1/2/4/8).
__device__ __forceinline__ float red16(float v) {
  v += __int_as_float(__builtin_amdgcn_ds_swizzle(__float_as_int(v), 0x041f));
  v += __int_as_float(__builtin_amdgcn_ds_swizzle(__float_as_int(v), 0x081f));
  v += __int_as_float(__builtin_amdgcn_ds_swizzle(__float_as_int(v), 0x101f));
  v += __int_as_float(__builtin_amdgcn_ds_swizzle(__float_as_int(v), 0x201f));
  return v;
}

// Stage one 16-byte chunk global -> LDS (async path on gfx1250 if available)
__device__ __forceinline__ void stage16(const void* g, void* l) {
#ifdef HAVE_ASYNC_LDS
  __builtin_amdgcn_global_load_async_to_lds_b128((gptr_v4i)g, (lptr_v4i)l, 0, 0);
#else
  *(uint4*)l = *(const uint4*)g;
#endif
}
__device__ __forceinline__ void stage_wait() {
#ifdef HAVE_ASYNC_LDS
  __builtin_amdgcn_s_wait_asynccnt(0);
#endif
  __syncthreads();
}

// ---- fragment loaders -------------------------------------------------------
// A-matrix 16x32 f16 layout (ISA 7.12.2): lane<16: row=lane, elems0-7=K0-7,
// elems8-15=K16-23 ; lane>=16: row=lane-16, elems0-7=K8-15, elems8-15=K24-31.
__device__ __forceinline__ v16h load_a_frag(const _Float16* __restrict__ xh,
                                            int row, int c, int hi) {
  const _Float16* p = xh + (size_t)row * HID + c * 32 + hi * 8;
  v8h lo = *(const v8h*)p;
  v8h hh = *(const v8h*)(p + 16);
  v16h a;
#pragma unroll
  for (int i = 0; i < 8; ++i) { a[i] = lo[i]; a[8 + i] = hh[i]; }
  return a;
}
// B fragment from LDS-staged pack: contiguous 16 halves per lane (ds_load).
__device__ __forceinline__ v16h lds_b_frag(const _Float16* W, int c, int nt,
                                           int lane) {
  return *(const v16h*)(W + (((c * 8 + nt) * 32 + lane) << 4));
}

// ---- prep kernels -----------------------------------------------------------
__global__ void k_cvt_x(const float* __restrict__ x, _Float16* __restrict__ xh,
                        int n) {
  int i = blockIdx.x * blockDim.x + threadIdx.x;
  if (i < n) xh[i] = (_Float16)x[i];
}

// Pack W[t][k][col] (kdim rows, 128 cols) into WMMA B-fragment layout:
// dst[((t*nchunks + c)*8 + nt)*32 + lane][e] with
// col = nt*16 + (lane&15), k = c*32 + (lane>>4)*8 + (e&7) + (e>>3)*16   (ISA B layout)
__global__ void k_pack_w(const float* __restrict__ W, _Float16* __restrict__ dst,
                         int kdim) {
  int gid = blockIdx.x * blockDim.x + threadIdx.x;
  int nchunks = kdim >> 5;
  int total = 3 * nchunks * 8 * 32 * 16;
  if (gid >= total) return;
  int e    = gid & 15;
  int lane = (gid >> 4) & 31;
  int nt   = (gid >> 9) & 7;
  int rest = gid >> 12;           // == t*nchunks + c
  int c    = rest % nchunks;
  int t    = rest / nchunks;
  int hi   = lane >> 4;
  int col  = nt * 16 + (lane & 15);
  int k    = c * 32 + hi * 8 + (e & 7) + ((e >> 3) << 4);
  dst[gid] = (_Float16)W[((size_t)t * kdim + k) * 128 + col];
}

// ---- pass 1: per-edge attention scores via WMMA (weights staged in LDS) -----
// grid: (ceil(tpt/8), 3)  block: 256 = 8 wave32, one 16-edge tile per wave
__global__ void __launch_bounds__(256) k_scores(
    const _Float16* __restrict__ xh,
    const _Float16* __restrict__ WqB, const _Float16* __restrict__ WkB,
    const int* __restrict__ a0, const int* __restrict__ a1,
    const int* __restrict__ a2, float* __restrict__ sbuf, int E) {
  __shared__ __align__(32) _Float16 lWq[4 * 8 * 32 * 16];   // 32 KB
  __shared__ __align__(32) _Float16 lWk[4 * 8 * 32 * 16];   // 32 KB
  int t   = blockIdx.y;
  int tid = threadIdx.x;
  {
    const uint4* gq = (const uint4*)(WqB + (size_t)t * 16384);
    const uint4* gk = (const uint4*)(WkB + (size_t)t * 16384);
    uint4* lq = (uint4*)lWq;
    uint4* lk = (uint4*)lWk;
    for (int i = tid; i < 2048; i += 256) {
      stage16(gq + i, lq + i);
      stage16(gk + i, lk + i);
    }
  }
  stage_wait();

  int tpt  = (E + 15) >> 4;
  int tile = blockIdx.x * 8 + (tid >> 5);
  if (tile >= tpt) return;                   // wave-uniform -> EXEC all 1s
  int e0 = tile << 4;
  const int* __restrict__ adj = (t == 0) ? a0 : (t == 1) ? a1 : a2;
  int lane = tid & 31;
  int r = lane & 15, hi = lane >> 4;
  int eL = e0 + r; if (eL >= E) eL = E - 1;
  int srcI = adj[2 * eL], tgtI = adj[2 * eL + 1];

  v16h aT[4], aS[4];
#pragma unroll
  for (int c = 0; c < 4; ++c) {
    aT[c] = load_a_frag(xh, tgtI, c, hi);
    aS[c] = load_a_frag(xh, srcI, c, hi);
  }
#pragma unroll
  for (int h = 0; h < 8; ++h) {              // head h == output n-tile h
    v8f q = {}; v8f k = {};
#pragma unroll
    for (int c = 0; c < 4; ++c) {
      q = __builtin_amdgcn_wmma_f32_16x16x32_f16(false, aT[c], false,
            lds_b_frag(lWq, c, h, lane), (short)0, q, false, false);
      k = __builtin_amdgcn_wmma_f32_16x16x32_f16(false, aS[c], false,
            lds_b_frag(lWk, c, h, lane), (short)0, k, false, false);
    }
    // s[e,h] = sum over 16 head-dims (cols) of q*k : reduce across 16 lanes
    float p[8];
#pragma unroll
    for (int i = 0; i < 8; ++i) p[i] = red16(q[i] * k[i]);
    if (r == 0) {
#pragma unroll
      for (int i = 0; i < 8; ++i) {
        int e = e0 + hi * 8 + i;
        if (e < E) sbuf[((size_t)t * E + e) * 8 + h] = p[i] * 0.25f; // *PHD^-0.5
      }
    }
  }
}

// ---- pass 2: segment max ----------------------------------------------------
__global__ void k_segmax(const float* __restrict__ sbuf,
                         const int* __restrict__ a0, const int* __restrict__ a1,
                         const int* __restrict__ a2,
                         unsigned* __restrict__ smax, int E) {
  int idx = blockIdx.x * blockDim.x + threadIdx.x;
  if (idx >= 3 * E * 8) return;
  int h  = idx & 7;
  int fe = idx >> 3;
  int t  = fe / E;
  int e  = fe - t * E;
  const int* adj = (t == 0) ? a0 : (t == 1) ? a1 : a2;
  int tgt = adj[2 * e + 1];
  atomicMax(&smax[tgt * 8 + h], ordEnc(sbuf[idx]));
}

// ---- pass 3: p = exp(s - max), segment sum ----------------------------------
__global__ void k_pun(float* __restrict__ sbuf,
                      const int* __restrict__ a0, const int* __restrict__ a1,
                      const int* __restrict__ a2,
                      const unsigned* __restrict__ smax,
                      float* __restrict__ sumexp, int E) {
  int idx = blockIdx.x * blockDim.x + threadIdx.x;
  if (idx >= 3 * E * 8) return;
  int h  = idx & 7;
  int fe = idx >> 3;
  int t  = fe / E;
  int e  = fe - t * E;
  const int* adj = (t == 0) ? a0 : (t == 1) ? a1 : a2;
  int tgt = adj[2 * e + 1];
  float mx = ordDec(smax[tgt * 8 + h]);
  float p  = __expf(sbuf[idx] - mx);
  sbuf[idx] = p;
  atomAddF(&sumexp[tgt * 8 + h], p);
}

// ---- pass 4: messages via WMMA (Wm staged in LDS) + weighted scatter-add ----
__global__ void __launch_bounds__(256) k_agg(
    const _Float16* __restrict__ xh, const _Float16* __restrict__ WmB,
    const float* __restrict__ bm,
    const int* __restrict__ a0, const int* __restrict__ a1,
    const int* __restrict__ a2,
    const float* __restrict__ pun, const float* __restrict__ sumexp,
    float* __restrict__ out, int E) {
  __shared__ __align__(32) _Float16 lWm[8 * 8 * 32 * 16];   // 64 KB
  int t   = blockIdx.y;
  int tid = threadIdx.x;
  {
    const uint4* gm = (const uint4*)(WmB + (size_t)t * 32768);
    uint4* lm = (uint4*)lWm;
    for (int i = tid; i < 4096; i += 256) stage16(gm + i, lm + i);
  }
  stage_wait();

  int tpt  = (E + 15) >> 4;
  int tile = blockIdx.x * 8 + (tid >> 5);
  if (tile >= tpt) return;
  int e0 = tile << 4;
  const int* __restrict__ adj = (t == 0) ? a0 : (t == 1) ? a1 : a2;
  int lane = tid & 31;
  int r = lane & 15, hi = lane >> 4;
  int eL = e0 + r; if (eL >= E) eL = E - 1;
  int srcI = adj[2 * eL], tgtI = adj[2 * eL + 1];

  // pair = [x_src (K 0..127) | x_tgt (K 128..255)]
  v16h aP[8];
#pragma unroll
  for (int c = 0; c < 4; ++c) aP[c]     = load_a_frag(xh, srcI, c, hi);
#pragma unroll
  for (int c = 0; c < 4; ++c) aP[4 + c] = load_a_frag(xh, tgtI, c, hi);

  int tgtM[8];
#pragma unroll
  for (int i = 0; i < 8; ++i) {
    int e = e0 + hi * 8 + i; if (e >= E) e = E - 1;
    tgtM[i] = adj[2 * e + 1];
  }
#pragma unroll
  for (int nt = 0; nt < 8; ++nt) {          // n-tile nt == head nt
    v8f acc = {};
#pragma unroll
    for (int c = 0; c < 8; ++c)
      acc = __builtin_amdgcn_wmma_f32_16x16x32_f16(false, aP[c], false,
              lds_b_frag(lWm, c, nt, lane), (short)0, acc, false, false);
    float b = bm[t * 128 + nt * 16 + r];    // lane col = r
#pragma unroll
    for (int i = 0; i < 8; ++i) {
      int e = e0 + hi * 8 + i;
      if (e < E) {
        float m = acc[i] + b;
        m = m > 0.f ? m : 0.f;
        float pu = pun[((size_t)t * E + e) * 8 + nt];
        float se = sumexp[tgtM[i] * 8 + nt];
        float prob = pu / (se > 0.f ? se : 1.f);
        atomAddF(out + (size_t)tgtM[i] * HID + nt * 16 + r, m * prob);
      }
    }
  }
}

// ---- host -------------------------------------------------------------------
extern "C" void kernel_launch(void* const* d_in, const int* in_sizes, int n_in,
                              void* d_out, int out_size, void* d_ws,
                              size_t ws_size, hipStream_t stream) {
  const float* x  = (const float*)d_in[0];
  const int*   a0 = (const int*)d_in[1];
  const int*   a1 = (const int*)d_in[2];
  const int*   a2 = (const int*)d_in[3];
  const float* Wq = (const float*)d_in[4];
  const float* Wk = (const float*)d_in[5];
  const float* Wm = (const float*)d_in[6];
  const float* bm = (const float*)d_in[7];
  int N = in_sizes[0] / HID;
  int E = in_sizes[1] / 2;

  char* ws = (char*)d_ws;
  size_t off = 0;
  _Float16* xh = (_Float16*)(ws + off);
  off += (size_t)N * HID * 2; off = (off + 255) & ~(size_t)255;
  const size_t szW128 = 3 * 4 * 8 * 32 * 16 * 2;   // 98304 B
  const size_t szW256 = 3 * 8 * 8 * 32 * 16 * 2;   // 196608 B
  _Float16* WqB = (_Float16*)(ws + off); off += szW128;
  _Float16* WkB = (_Float16*)(ws + off); off += szW128;
  _Float16* WmB = (_Float16*)(ws + off); off += szW256;
  off = (off + 255) & ~(size_t)255;
  float*    sbuf   = (float*)(ws + off); off += (size_t)3 * E * 8 * 4;
  unsigned* smax   = (unsigned*)(ws + off); off += (size_t)N * 8 * 4;
  float*    sumexp = (float*)(ws + off); off += (size_t)N * 8 * 4;

  (void)hipMemsetAsync(d_out, 0, (size_t)N * HID * 4, stream);
  (void)hipMemsetAsync(smax, 0, (size_t)N * 8 * 4, stream); // ordEnc: 0 < real
  (void)hipMemsetAsync(sumexp, 0, (size_t)N * 8 * 4, stream);

  int nx = N * HID;
  k_cvt_x<<<(nx + 255) / 256, 256, 0, stream>>>(x, xh, nx);
  k_pack_w<<<(3 * 4 * 8 * 32 * 16 + 255) / 256, 256, 0, stream>>>(Wq, WqB, 128);
  k_pack_w<<<(3 * 4 * 8 * 32 * 16 + 255) / 256, 256, 0, stream>>>(Wk, WkB, 128);
  k_pack_w<<<(3 * 8 * 8 * 32 * 16 + 255) / 256, 256, 0, stream>>>(Wm, WmB, 256);

  int tpt = (E + 15) / 16;
  dim3 grid((tpt + 7) / 8, 3);
  k_scores<<<grid, 256, 0, stream>>>(xh, WqB, WkB, a0, a1, a2, sbuf, E);

  int tot = 3 * E * 8;
  k_segmax<<<(tot + 255) / 256, 256, 0, stream>>>(sbuf, a0, a1, a2, smax, E);
  k_pun<<<(tot + 255) / 256, 256, 0, stream>>>(sbuf, a0, a1, a2, smax, sumexp, E);

  k_agg<<<grid, 256, 0, stream>>>(xh, WmB, bm, a0, a1, a2, sbuf, sumexp,
                                  (float*)d_out, E);
}
